// MultiHeadAttention_13451837571761
// MI455X (gfx1250) — matrix operand order repaired and checked by reading, not compiled, not run on hardware
//
#include <hip/hip_runtime.h>
#include <hip/hip_bf16.h>
#include <math.h>

// ---------------- problem constants ----------------
constexpr int SEQ    = 1024;
constexpr int DMODEL = 512;
constexpr int NHEAD  = 8;
constexpr int DEP    = 64;
constexpr int BATCH  = 4;
constexpr int ROWS   = BATCH * SEQ;        // 4096
constexpr int BH     = BATCH * NHEAD;      // 32

constexpr int BK    = 64;   // K slab staged per barrier round
constexpr int BPAD  = 72;   // padded LDS row (bf16 elems): 144B stride -> conflict-free frags
constexpr int SLABE = 64 * BPAD;  // elems per LDS slab

typedef __attribute__((ext_vector_type(16))) __bf16 v16bf;
typedef __attribute__((ext_vector_type(8)))  float  v8f;

union FragU { v16bf v; uint4 q[2]; };

#if defined(__has_builtin)
#  if __has_builtin(__builtin_amdgcn_tensor_load_to_lds)
#    define HAVE_TDM 1
#  endif
#endif
#ifndef HAVE_TDM
#  define HAVE_TDM 0
#endif

__device__ __forceinline__ unsigned lds_off_of(const void* p) {
  return (unsigned)(unsigned long long)p;   // generic addr[31:0] == LDS offset (ISA 10.2)
}

// ---------------- CDNA5 async copy: global -> LDS (ASYNCcnt tracked) ----------------
__device__ __forceinline__ void async_b128_to_lds(const __bf16* gsrc, __bf16* lds_dst) {
  unsigned loff = lds_off_of(lds_dst);
  asm volatile("global_load_async_to_lds_b128 %0, %1, off"
               :: "v"(loff), "v"(gsrc) : "memory");
}
__device__ __forceinline__ void wait_async_all() {
  asm volatile("s_wait_asynccnt 0" ::: "memory");
}

// ---------------- CDNA5 Tensor Data Mover: 64x64 bf16 tile -> LDS (TENSORcnt) ----------------
#if HAVE_TDM
typedef unsigned int tdm_u32x4 __attribute__((ext_vector_type(4)));
typedef int          tdm_i32x4 __attribute__((ext_vector_type(4)));
typedef int          tdm_i32x8 __attribute__((ext_vector_type(8)));

// D# per ISA 8.3-8.5: 2D tile, tile_dim0=64 elems (128B rows), tile_dim1=64 rows,
// data_size=2B, LDS padding: every 32 DWORDs (one row) insert 4 DWORDs -> 144B stride (==BPAD).
__device__ __forceinline__ void tdm_load_tile_64x64(const __bf16* gsrc, unsigned lds_addr,
                                                    unsigned ld /*row stride, elems*/,
                                                    unsigned rows_total) {
  unsigned long long ga = (unsigned long long)gsrc;
  tdm_u32x4 g0;
  g0.x = 1u;                                                   // count=1 (valid), no gather
  g0.y = lds_addr;                                             // lds_addr[31:0]
  g0.z = (unsigned)ga;                                         // global_addr[31:0]
  g0.w = (unsigned)((ga >> 32) & 0x1FFFFFFull) | (2u << 30);   // global_addr[56:32] | type=2
  tdm_i32x8 g1;
  g1[0] = (1 << 16)            // data_size = 1 -> 2 bytes
        | (1 << 20)            // pad_enable
        | (4 << 22)            // pad_interval: code 4 -> every 32 DWORDs
        | (3 << 25);           // pad_amount:   code 3 -> 4 DWORDs (16B)
  g1[1] = (int)((ld & 0xFFFFu) << 16);                               // tensor_dim0[15:0]
  g1[2] = (int)(((ld >> 16) & 0xFFFFu) | ((rows_total & 0xFFFFu) << 16)); // td0 hi | td1 lo
  g1[3] = (int)(((rows_total >> 16) & 0xFFFFu) | (64u << 16));       // td1 hi | tile_dim0=64
  g1[4] = 64;                                                        // tile_dim1=64, tile_dim2=0
  g1[5] = (int)ld;                                                   // tensor_dim0_stride lo32
  g1[6] = 0;                                                         // stride hi16 | dim1_stride lo
  g1[7] = 0;
  tdm_i32x4 z4 = {0, 0, 0, 0};
#if __clang_major__ >= 23
  tdm_i32x8 z8 = {0, 0, 0, 0, 0, 0, 0, 0};
  __builtin_amdgcn_tensor_load_to_lds(g0, g1, z4, z4, z8, 0);
#else
  __builtin_amdgcn_tensor_load_to_lds(g0, g1, z4, z4, 0);
#endif
}
#endif  // HAVE_TDM

// ---------------- WMMA fragment loaders ----------------
// A-matrix 16x32 bf16 (ISA 7.12.2): lane L, M = L%16;
//   L<16 : K = k0..k0+7  (v0..3), k0+16..k0+23 (v4..7)
//   L>=16: K = k0+8..+15 (v0..3), k0+24..+31  (v4..7)
__device__ __forceinline__ v16bf load_a_frag(const __bf16* __restrict__ A, int lda,
                                             int m0, int k0) {
  const int lane = threadIdx.x & 31;
  const __bf16* p = A + (long)(m0 + (lane & 15)) * lda + k0 + ((lane >> 4) << 3);
  FragU f;
  f.q[0] = *(const uint4*)p;          // 8 bf16: K chunk lo
  f.q[1] = *(const uint4*)(p + 16);   // 8 bf16: K chunk +16
  return f.v;
}
// B-matrix 32x16 bf16 from LDS tile bsm[n][k] (row stride BPAD): lane L, N = L%16;
//   L<16 : K = k0..k0+7, k0+16..k0+23 ; L>=16: K = k0+8..k0+15, k0+24..k0+31
__device__ __forceinline__ v16bf load_b_frag_lds(const __bf16* bsm, int n0l, int k0l) {
  const int lane = threadIdx.x & 31;
  const __bf16* p = bsm + (n0l + (lane & 15)) * BPAD + k0l + ((lane >> 4) << 3);
  FragU f;
  f.q[0] = *(const uint4*)p;
  f.q[1] = *(const uint4*)(p + 16);
  return f.v;
}

// one K-slab of WMMA work for a wave's 16x64 tile; B comes from LDS slab `cur`
__device__ __forceinline__ void wmma_slab(const __bf16* __restrict__ A, int lda,
                                          int m0, int k0, const __bf16* cur, v8f acc[4]) {
#pragma unroll
  for (int kk = 0; kk < BK; kk += 32) {
    v16bf af = load_a_frag(A, lda, m0, k0 + kk);
    v16bf b0 = load_b_frag_lds(cur,  0, kk);
    v16bf b1 = load_b_frag_lds(cur, 16, kk);
    v16bf b2 = load_b_frag_lds(cur, 32, kk);
    v16bf b3 = load_b_frag_lds(cur, 48, kk);
    acc[0] = __builtin_amdgcn_wmma_f32_16x16x32_bf16(false, af, false, b0, (short)0, acc[0], false, false);
    acc[1] = __builtin_amdgcn_wmma_f32_16x16x32_bf16(false, af, false, b1, (short)0, acc[1], false, false);
    acc[2] = __builtin_amdgcn_wmma_f32_16x16x32_bf16(false, af, false, b2, (short)0, acc[2], false, false);
    acc[3] = __builtin_amdgcn_wmma_f32_16x16x32_bf16(false, af, false, b3, (short)0, acc[3], false, false);
  }
}

// ---------------- workgroup-cooperative GEMMs ----------------
// Block: 8 wave32; all waves share Bt columns [n0,n0+64); wave computes 16(M)x64(N).

// (a) per-lane async-to-LDS staging, single buffer (used for K=64 scores GEMM)
__device__ __forceinline__ void gemm16x64_coop(const __bf16* __restrict__ A, int lda,
                                               const __bf16* __restrict__ Bt, int ldb,
                                               int m0, int n0, int K, v8f acc[4],
                                               __bf16* bsm /* [64][BPAD] */) {
  const int tid = threadIdx.x;
  for (int k0 = 0; k0 < K; k0 += BK) {
#pragma unroll
    for (int tr = 0; tr < 2; ++tr) {          // 512 x 16B transfers, 2 per thread
      int t = tid + tr * 256;
      int r = t >> 3, c = t & 7;
      async_b128_to_lds(Bt + (long)(n0 + r) * ldb + k0 + c * 8,
                        bsm + r * BPAD + c * 8);
    }
    wait_async_all();
    __syncthreads();
    wmma_slab(A, lda, m0, k0, bsm, acc);
    __syncthreads();
  }
}

// (b) TDM-staged, double buffered (used for K>=512 GEMMs); falls back to (a) w/o TDM
__device__ __forceinline__ void gemm16x64_tdm(const __bf16* __restrict__ A, int lda,
                                              const __bf16* __restrict__ Bt, int ldb,
                                              int m0, int n0, int K, unsigned nrows,
                                              v8f acc[4], __bf16* bsm /* [2][64*BPAD] */) {
#if HAVE_TDM
  const __bf16* tile0 = Bt + (long)n0 * ldb;
  const int nslab = K / BK;
  const bool issuer = (threadIdx.x >> 5) == 0;   // wave 0 only (TDM ignores EXEC)
  if (issuer)
    tdm_load_tile_64x64(tile0, lds_off_of(bsm), (unsigned)ldb, nrows);
  for (int i = 0; i < nslab; ++i) {
    if (issuer) __builtin_amdgcn_s_wait_tensorcnt(0);   // slab i landed in LDS
    __syncthreads();                                    // publish; all done reading buf^1
    if (issuer && (i + 1 < nslab))                      // overlap: fetch slab i+1
      tdm_load_tile_64x64(tile0 + (i + 1) * BK,
                          lds_off_of(bsm + ((i + 1) & 1) * SLABE),
                          (unsigned)ldb, nrows);
    wmma_slab(A, lda, m0, i * BK, bsm + (i & 1) * SLABE, acc);
  }
#else
  gemm16x64_coop(A, lda, Bt, ldb, m0, n0, K, acc, bsm);
#endif
}

// C/D 16x16 layout: lane L holds N = L%16; VGPR r holds M = r + (L>=16 ? 8 : 0)
#define TILE_M(r)  (m0 + ((lane >> 4) << 3) + (r))
#define TILE_N(t)  (n0 + (t) * 16 + (lane & 15))

// ---------------- small helpers ----------------
__device__ __forceinline__ void atomicMaxF(float* addr, float val) {
  if (val >= 0.f) atomicMax((int*)addr, __float_as_int(val));
  else            atomicMin((unsigned int*)addr, __float_as_uint(val));
}
__device__ __forceinline__ float waveMax(float v) {
#pragma unroll
  for (int off = 16; off > 0; off >>= 1) v = fmaxf(v, __shfl_xor(v, off, 32));
  return v;
}
__device__ __forceinline__ float waveSum(float v) {
#pragma unroll
  for (int off = 16; off > 0; off >>= 1) v += __shfl_xor(v, off, 32);
  return v;
}

// ---------------- prep kernels ----------------
__global__ void init_red_kernel(float* red_max, float* red_sum) {
  int i = threadIdx.x;
  if (i < BH) { red_max[i] = -INFINITY; red_sum[i] = 0.f; }
}

__global__ void cvt_bf16_kernel(const float* __restrict__ q, const float* __restrict__ k,
                                const float* __restrict__ v,
                                __bf16* qb, __bf16* kb, __bf16* vb, int n) {
  int i = blockIdx.x * blockDim.x + threadIdx.x;
  if (i < n) { qb[i] = (__bf16)q[i]; kb[i] = (__bf16)k[i]; vb[i] = (__bf16)v[i]; }
}

// transpose 512x512 fp32 W[in][out] -> bf16 Wt[out][in], LDS tiled
__global__ void transpose_w_kernel(const float* __restrict__ w0, const float* __restrict__ w1,
                                   const float* __restrict__ w2, const float* __restrict__ w3,
                                   __bf16* t0, __bf16* t1, __bf16* t2, __bf16* t3) {
  __shared__ float tile[32][33];
  const float* w = (blockIdx.z == 0) ? w0 : (blockIdx.z == 1) ? w1 : (blockIdx.z == 2) ? w2 : w3;
  __bf16*      t = (blockIdx.z == 0) ? t0 : (blockIdx.z == 1) ? t1 : (blockIdx.z == 2) ? t2 : t3;
  int i0 = blockIdx.y * 32, o0 = blockIdx.x * 32;
  for (int r = threadIdx.y; r < 32; r += 8)
    tile[r][threadIdx.x] = w[(long)(i0 + r) * DMODEL + o0 + threadIdx.x];
  __syncthreads();
  for (int r = threadIdx.y; r < 32; r += 8)
    t[(long)(o0 + r) * DMODEL + i0 + threadIdx.x] = (__bf16)tile[threadIdx.x][r];
}

// ---------------- QKV projections ----------------
// mode 0/1: dst = head-split [b][h][s][d] ; mode 2: dst = transposed [b][h][d][s]
__global__ void proj_kernel(const __bf16* __restrict__ X, const __bf16* __restrict__ Wt,
                            const float* __restrict__ bias, __bf16* __restrict__ dst, int mode) {
  __shared__ __align__(16) __bf16 bsm[2 * SLABE];
  const int lane = threadIdx.x & 31, w = threadIdx.x >> 5;
  const int m0 = blockIdx.y * 128 + w * 16;
  const int n0 = blockIdx.x * 64;
  v8f acc[4] = {};
  gemm16x64_tdm(X, DMODEL, Wt, DMODEL, m0, n0, DMODEL, DMODEL, acc, bsm);
#pragma unroll
  for (int t = 0; t < 4; ++t) {
    const float* ap = (const float*)&acc[t];
#pragma unroll
    for (int r = 0; r < 8; ++r) {
      int m = TILE_M(r), n = TILE_N(t);
      float vv = ap[r] + bias[n];
      int b = m >> 10, s = m & (SEQ - 1);
      int h = n >> 6,  d = n & (DEP - 1);
      long idx = (mode == 2)
          ? ((long)(b * NHEAD + h) * DEP + d) * SEQ + s
          : ((long)(b * NHEAD + h) * SEQ + s) * DEP + d;
      dst[idx] = (__bf16)vv;
    }
  }
}

// ---------------- logits (scores -> masked/augmented) ----------------
__device__ __forceinline__ void logits_tile(const __bf16* __restrict__ kh,
                                            const __bf16* __restrict__ qh,
                                            const float* __restrict__ mk,
                                            const float* __restrict__ thb,
                                            const float* __restrict__ tpi,
                                            float at, float hb, float pi,
                                            int m0, int n0, float out[4][8],
                                            __bf16* bsm) {
  v8f acc[4] = {};
  // scores[i][j] = sum_d kh[i,d] * qh[j,d]   (reference swaps q/k)
  gemm16x64_coop(kh, DEP, qh, DEP, m0, n0, DEP, acc, bsm);
  const int lane = threadIdx.x & 31;
#pragma unroll
  for (int t = 0; t < 4; ++t) {
    const float* ap = (const float*)&acc[t];
#pragma unroll
    for (int r = 0; r < 8; ++r) {
      int m = TILE_M(r), n = TILE_N(t);
      long e = (long)m * SEQ + n;
      float sc = ap[r] * 0.125f + mk[e] * (-1e9f);   // 1/sqrt(64) = 0.125
      out[t][r] = sc * at + hb * thb[e] + pi * tpi[e];
    }
  }
}

__global__ void logits_max_kernel(const __bf16* __restrict__ qh, const __bf16* __restrict__ kh,
                                  const float* __restrict__ mask, const float* __restrict__ thb,
                                  const float* __restrict__ tpi,
                                  const float* __restrict__ aug_hb, const float* __restrict__ aug_pi,
                                  const float* __restrict__ aug_at, const int* __restrict__ bn,
                                  float* red_max) {
  __shared__ __align__(16) __bf16 bsm[SLABE];
  const int bh = blockIdx.z, b = bh >> 3;
  const int blk = bn[0];
  const float at = aug_at[blk], hb = aug_hb[blk], pi = aug_pi[blk];
  const int lane = threadIdx.x & 31, w = threadIdx.x >> 5;
  const int m0 = blockIdx.y * 128 + w * 16, n0 = blockIdx.x * 64;
  float out[4][8];
  logits_tile(kh + (long)bh * SEQ * DEP, qh + (long)bh * SEQ * DEP,
              mask + (long)b * SEQ * SEQ, thb, tpi, at, hb, pi, m0, n0, out, bsm);
  float mx = -INFINITY;
#pragma unroll
  for (int t = 0; t < 4; ++t)
#pragma unroll
    for (int r = 0; r < 8; ++r) mx = fmaxf(mx, out[t][r]);
  mx = waveMax(mx);
  __shared__ float sm[8];
  if (lane == 0) sm[w] = mx;
  __syncthreads();
  if (threadIdx.x == 0) {
    float m2 = sm[0];
#pragma unroll
    for (int i = 1; i < 8; ++i) m2 = fmaxf(m2, sm[i]);
    atomicMaxF(&red_max[bh], m2);
  }
}

__global__ void exp_sum_kernel(const __bf16* __restrict__ qh, const __bf16* __restrict__ kh,
                               const float* __restrict__ mask, const float* __restrict__ thb,
                               const float* __restrict__ tpi,
                               const float* __restrict__ aug_hb, const float* __restrict__ aug_pi,
                               const float* __restrict__ aug_at, const int* __restrict__ bn,
                               const float* __restrict__ red_max,
                               float* __restrict__ attn, __bf16* __restrict__ e_bf,
                               float* red_sum) {
  __shared__ __align__(16) __bf16 bsm[SLABE];
  const int bh = blockIdx.z, b = bh >> 3;
  const int blk = bn[0];
  const float at = aug_at[blk], hb = aug_hb[blk], pi = aug_pi[blk];
  const int lane = threadIdx.x & 31, w = threadIdx.x >> 5;
  const int m0 = blockIdx.y * 128 + w * 16, n0 = blockIdx.x * 64;
  float out[4][8];
  logits_tile(kh + (long)bh * SEQ * DEP, qh + (long)bh * SEQ * DEP,
              mask + (long)b * SEQ * SEQ, thb, tpi, at, hb, pi, m0, n0, out, bsm);
  const float mx = red_max[bh];
  float* ap_out   = attn + (long)bh * SEQ * SEQ;
  __bf16* ep_out  = e_bf + (long)bh * SEQ * SEQ;
  float lsum = 0.f;
#pragma unroll
  for (int t = 0; t < 4; ++t)
#pragma unroll
    for (int r = 0; r < 8; ++r) {
      int m = TILE_M(r), n = TILE_N(t);
      float e = __expf(out[t][r] - mx);
      long idx = (long)m * SEQ + n;
      ap_out[idx] = e;           // unnormalized; normalized in pass C
      ep_out[idx] = (__bf16)e;   // bf16 copy for ctx GEMM
      lsum += e;
    }
  lsum = waveSum(lsum);
  __shared__ float sm[8];
  if (lane == 0) sm[w] = lsum;
  __syncthreads();
  if (threadIdx.x == 0) {
    float s2 = 0.f;
#pragma unroll
    for (int i = 0; i < 8; ++i) s2 += sm[i];
    atomicAdd(&red_sum[bh], s2);
  }
}

__global__ void normalize_kernel(float* __restrict__ attn, const float* __restrict__ red_sum) {
  long i4 = (long)blockIdx.x * blockDim.x + threadIdx.x;   // float4 index
  int bh = (int)(i4 >> 18);                                // (S*S)/4 = 2^18 per head
  float inv = 1.0f / red_sum[bh];
  float4* p = (float4*)attn + i4;
  float4 vv = *p;
  vv.x *= inv; vv.y *= inv; vv.z *= inv; vv.w *= inv;
  *p = vv;
}

// ---------------- ctx = attn @ vh (per head), then output projection ----------------
__global__ void ctx_kernel(const __bf16* __restrict__ e_bf, const __bf16* __restrict__ vhT,
                           const float* __restrict__ red_sum, __bf16* __restrict__ ctx_bf) {
  __shared__ __align__(16) __bf16 bsm[2 * SLABE];
  const int bh = blockIdx.z, b = bh >> 3, h = bh & 7;
  const int lane = threadIdx.x & 31, w = threadIdx.x >> 5;
  const int m0 = blockIdx.y * 128 + w * 16, n0 = 0;
  v8f acc[4] = {};
  gemm16x64_tdm(e_bf + (long)bh * SEQ * SEQ, SEQ, vhT + (long)bh * DEP * SEQ, SEQ,
                m0, n0, SEQ, DEP, acc, bsm);
  const float inv = 1.0f / red_sum[bh];
#pragma unroll
  for (int t = 0; t < 4; ++t) {
    const float* ap = (const float*)&acc[t];
#pragma unroll
    for (int r = 0; r < 8; ++r) {
      int m = TILE_M(r), n = TILE_N(t);     // m = s index, n = d index
      ctx_bf[((long)(b * SEQ + m)) * DMODEL + h * DEP + n] = (__bf16)(ap[r] * inv);
    }
  }
}

__global__ void outproj_kernel(const __bf16* __restrict__ ctx_bf, const __bf16* __restrict__ woT,
                               const float* __restrict__ bias, float* __restrict__ out) {
  __shared__ __align__(16) __bf16 bsm[2 * SLABE];
  const int lane = threadIdx.x & 31, w = threadIdx.x >> 5;
  const int m0 = blockIdx.y * 128 + w * 16;
  const int n0 = blockIdx.x * 64;
  v8f acc[4] = {};
  gemm16x64_tdm(ctx_bf, DMODEL, woT, DMODEL, m0, n0, DMODEL, DMODEL, acc, bsm);
#pragma unroll
  for (int t = 0; t < 4; ++t) {
    const float* ap = (const float*)&acc[t];
#pragma unroll
    for (int r = 0; r < 8; ++r) {
      int m = TILE_M(r), n = TILE_N(t);
      out[(long)m * DMODEL + n] = ap[r] + bias[n];
    }
  }
}

// ---------------- launcher ----------------
extern "C" void kernel_launch(void* const* d_in, const int* in_sizes, int n_in,
                              void* d_out, int out_size, void* d_ws, size_t ws_size,
                              hipStream_t stream) {
  const float* v_in  = (const float*)d_in[0];
  const float* q_in  = (const float*)d_in[1];
  const float* k_in  = (const float*)d_in[2];
  const float* wq_w  = (const float*)d_in[3];
  const float* wq_b  = (const float*)d_in[4];
  const float* wk_w  = (const float*)d_in[5];
  const float* wk_b  = (const float*)d_in[6];
  const float* wv_w  = (const float*)d_in[7];
  const float* wv_b  = (const float*)d_in[8];
  const float* wo_w  = (const float*)d_in[9];
  const float* wo_b  = (const float*)d_in[10];
  const float* aug_hb = (const float*)d_in[11];
  const float* aug_pi = (const float*)d_in[12];
  const float* aug_at = (const float*)d_in[13];
  const float* thb    = (const float*)d_in[14];
  const float* tpi    = (const float*)d_in[15];
  const float* mask   = (const float*)d_in[16];
  const int*   bn     = (const int*)d_in[17];

  float* out  = (float*)d_out;                       // (B,S,DMODEL)
  float* attn = (float*)d_out + (long)ROWS * DMODEL; // (B,H,S,S)

  // workspace layout
  const size_t MB = 1u << 20;
  char* ws = (char*)d_ws;
  __bf16* q_bf  = (__bf16*)(ws + 0 * MB);
  __bf16* k_bf  = (__bf16*)(ws + 4 * MB);
  __bf16* v_bf  = (__bf16*)(ws + 8 * MB);
  __bf16* wqT   = (__bf16*)(ws + 12 * MB);
  __bf16* wkT   = (__bf16*)(ws + 12 * MB + 512 * 1024);
  __bf16* wvT   = (__bf16*)(ws + 13 * MB);
  __bf16* woT   = (__bf16*)(ws + 13 * MB + 512 * 1024);
  __bf16* qh    = (__bf16*)(ws + 14 * MB);   // [b][h][s][d]
  __bf16* kh    = (__bf16*)(ws + 18 * MB);   // [b][h][s][d]
  __bf16* vhT   = (__bf16*)(ws + 22 * MB);   // [b][h][d][s]
  __bf16* ctxb  = (__bf16*)(ws + 26 * MB);   // [b*s][dmodel]
  float*  red   = (float*)(ws + 30 * MB);
  float*  red_max = red;
  float*  red_sum = red + BH;
  __bf16* e_bf  = (__bf16*)(ws + 30 * MB + 4096); // [b][h][s][s] bf16, 64 MB

  const int NELEM = ROWS * DMODEL; // 2,097,152

  init_red_kernel<<<1, 64, 0, stream>>>(red_max, red_sum);
  cvt_bf16_kernel<<<(NELEM + 255) / 256, 256, 0, stream>>>(q_in, k_in, v_in,
                                                           q_bf, k_bf, v_bf, NELEM);
  transpose_w_kernel<<<dim3(16, 16, 4), dim3(32, 8), 0, stream>>>(
      wq_w, wk_w, wv_w, wo_w, wqT, wkT, wvT, woT);

  // projections: grid = (N/64, M/128) = (8, 32), 256 threads = 8 wave32
  proj_kernel<<<dim3(8, 32), 256, 0, stream>>>(q_bf, wqT, wq_b, qh, 0);
  proj_kernel<<<dim3(8, 32), 256, 0, stream>>>(k_bf, wkT, wk_b, kh, 1);
  proj_kernel<<<dim3(8, 32), 256, 0, stream>>>(v_bf, wvT, wv_b, vhT, 2);

  // joint softmax over (S,S): pass1 max (recompute), pass2 exp+sum, pass3 normalize
  logits_max_kernel<<<dim3(16, 8, BH), 256, 0, stream>>>(
      qh, kh, mask, thb, tpi, aug_hb, aug_pi, aug_at, bn, red_max);
  exp_sum_kernel<<<dim3(16, 8, BH), 256, 0, stream>>>(
      qh, kh, mask, thb, tpi, aug_hb, aug_pi, aug_at, bn, red_max, attn, e_bf, red_sum);
  normalize_kernel<<<(BH * SEQ * SEQ / 4) / 256, 256, 0, stream>>>(attn, red_sum);

  ctx_kernel<<<dim3(1, 8, BH), 256, 0, stream>>>(e_bf, vhT, red_sum, ctxb);
  outproj_kernel<<<dim3(8, 32), 256, 0, stream>>>(ctxb, woT, wo_b, out);
}
